// Model_34402688041398
// MI455X (gfx1250) — matrix operand order repaired and checked by reading, not compiled, not run on hardware
//
#include <hip/hip_runtime.h>
#include <hip/hip_bf16.h>

typedef __attribute__((ext_vector_type(16))) _Float16 v16h;
typedef __attribute__((ext_vector_type(8)))  _Float16 v8h;
typedef __attribute__((ext_vector_type(8)))  float    v8f;

static constexpr int   Bn   = 4096;
static constexpr int   Cc   = 10;
static constexpr int   DinC = 3072;   // 3*32*32
static constexpr int   Dd   = 256;
static constexpr int   Kc   = 512;
static constexpr float BETA = 0.25f;

// ---------------------------------------------------------------------------
// Fragment helpers (wave32, v_wmma_f32_16x16x32_f16 layouts per CDNA5 ISA 7.12.2)
// A 16x32 f16: lanes 0-15 -> M=lane, K {0..7,16..23}; lanes 16-31 -> M=lane-16, K {8..15,24..31}
// B 32x16 f16: lane n=lane&15 holds column n; h[0..15] = K (lane<16 ? {0..7,16..23} : {8..15,24..31})
// C/D f32:     vgpr r, lane -> M = r + (lane>=16?8:0), N = lane&15
// ---------------------------------------------------------------------------

__device__ inline v16h load_a_frag_f32(const float* rowp, int k0, int half) {
    const float* p = rowp + k0 + half * 8;
    v16h a;
#pragma unroll
    for (int j = 0; j < 8; ++j) {
        a[j]     = (_Float16)p[j];
        a[8 + j] = (_Float16)p[16 + j];
    }
    return a;
}

__device__ inline v16h load_a_frag_f16(const _Float16* rowp, int k0, int half) {
    const _Float16* p = rowp + k0 + half * 8;
    v8h lo = *(const v8h*)(p);
    v8h hi = *(const v8h*)(p + 16);
    v16h a;
#pragma unroll
    for (int j = 0; j < 8; ++j) { a[j] = lo[j]; a[8 + j] = hi[j]; }
    return a;
}

// B fragment from a column-major (K contiguous) f16 matrix: two 16B loads per lane, K {8h..8h+7, 16+8h..16+8h+7}
__device__ inline v16h load_b_frag_t(const _Float16* colmaj, int n, int ldK,
                                     int k0, int half) {
    const _Float16* p = colmaj + (size_t)n * ldK + k0 + half * 8;
    v8h lo = *(const v8h*)(p);
    v8h hi = *(const v8h*)(p + 16);
    v16h b;
#pragma unroll
    for (int j = 0; j < 8; ++j) { b[j] = lo[j]; b[8 + j] = hi[j]; }
    return b;
}

// ---------------------------------------------------------------------------
// 0) counts <- 0
__global__ void k_zero_counts(int* counts) {
    if (threadIdx.x < Cc) counts[threadIdx.x] = 0;
}

// 1) bucket rows by label (order within a class does not affect any value)
__global__ void k_scatter_rows(const int* __restrict__ label, int* counts,
                               int* rowlist) {
    int b = blockIdx.x * blockDim.x + threadIdx.x;
    if (b < Bn) {
        int c   = label[b];
        int pos = atomicAdd(&counts[c], 1);
        rowlist[c * Bn + pos] = b;
    }
}

// 2) codebook f32 -> f16 copy ([n][k] is already the B layout) + squared norms
__global__ __launch_bounds__(64) void k_prep_cb(const float* __restrict__ cb,
                                                _Float16* __restrict__ cbh,
                                                float* __restrict__ cbn) {
    __shared__ float red[64];
    int r = blockIdx.x, tid = threadIdx.x;
    float s = 0.f;
    for (int j = tid; j < Dd; j += 64) {
        float v = cb[(size_t)r * Dd + j];
        cbh[(size_t)r * Dd + j] = (_Float16)v;
        s += v * v;
    }
    red[tid] = s;
    __syncthreads();
    if (tid == 0) {
        float t = 0.f;
        for (int i = 0; i < 64; ++i) t += red[i];   // fixed order
        cbn[r] = t;
    }
}

// 2b) tiled transpose: f32 [R x C] row-major -> f16 [C x R] (K becomes contiguous)
__global__ __launch_bounds__(256) void k_transpose_f32_f16(
    const float* __restrict__ src, _Float16* __restrict__ dst,
    int R, int C, size_t batch_elems) {
    __shared__ _Float16 tile[32][34];     // 34 halfs = 17 dwords: conflict-free
    const float* s = src + (size_t)blockIdx.z * batch_elems;
    _Float16*    d = dst + (size_t)blockIdx.z * batch_elems;
    int c0 = blockIdx.x * 32, r0 = blockIdx.y * 32;
    int tx = threadIdx.x & 31, ty = threadIdx.x >> 5;   // 32 x 8
#pragma unroll
    for (int i = 0; i < 32; i += 8)
        tile[ty + i][tx] = (_Float16)s[(size_t)(r0 + ty + i) * C + c0 + tx];
    __syncthreads();
#pragma unroll
    for (int i = 0; i < 32; i += 8)
        d[(size_t)(c0 + ty + i) * R + r0 + tx] = tile[tx][ty + i];
}

// 3) label-routed encoder: z[rows] = x[rows] @ enc_W[c] + enc_b[c]
//    16 gathered rows x D=256 per block; 4 waves x 4 n-tiles; B streamed from L2
__global__ __launch_bounds__(128) void k_encode(
    const float* __restrict__ x, const _Float16* __restrict__ encWt,
    const float* __restrict__ encb, const int* __restrict__ counts,
    const int* __restrict__ rowlist, float* __restrict__ z,
    _Float16* __restrict__ zh) {
    constexpr int TILES = Bn / 16;            // 256 tiles per class (padded)
    int c = blockIdx.x / TILES;
    int t = blockIdx.x % TILES;
    int cnt = counts[c];
    if (t * 16 >= cnt) return;

    __shared__ int rows[16];
    int tid  = threadIdx.x;
    int wave = tid >> 5, lane = tid & 31;
    int half = lane >> 4, nlane = lane & 15;

    if (tid < 16) {
        int i = t * 16 + tid;
        rows[tid] = rowlist[c * Bn + (i < cnt ? i : cnt - 1)];
    }
    __syncthreads();

    const float*    arow = x + (size_t)rows[nlane] * DinC;    // A row, M = lane&15
    const _Float16* Wt   = encWt + (size_t)c * DinC * Dd;     // [n][k], k contiguous

    v8f acc[4] = {};
    for (int k0 = 0; k0 < DinC; k0 += 32) {
        if (k0 + 32 < DinC) {
            __builtin_prefetch(arow + k0 + 32, 0, 0);
#pragma unroll
            for (int nt = 0; nt < 4; ++nt)
                __builtin_prefetch(
                    Wt + (size_t)(wave * 64 + nt * 16 + nlane) * DinC + k0 + 32, 0, 0);
        }
        v16h a = load_a_frag_f32(arow, k0, half);
#pragma unroll
        for (int nt = 0; nt < 4; ++nt) {
            v16h b = load_b_frag_t(Wt, wave * 64 + nt * 16 + nlane, DinC, k0, half);
            acc[nt] = __builtin_amdgcn_wmma_f32_16x16x32_f16(
                false, a, false, b, (short)0, acc[nt], false, false);
        }
    }
#pragma unroll
    for (int nt = 0; nt < 4; ++nt) {
        int col    = wave * 64 + nt * 16 + nlane;
        float bias = encb[c * Dd + col];
#pragma unroll
        for (int r = 0; r < 8; ++r) {
            int m = r + half * 8;
            if (t * 16 + m < cnt) {
                int row = rows[m];
                float v = acc[nt][r] + bias;
                z[(size_t)row * Dd + col]  = v;
                zh[(size_t)row * Dd + col] = (_Float16)v;
            }
        }
    }
}

// 4) VQ: scores = ||c_k||^2 - 2 z.c_k via WMMA, fixed-order argmin, per-sample loss
__global__ __launch_bounds__(128) void k_vq(
    const float* __restrict__ z, const _Float16* __restrict__ zh,
    const _Float16* __restrict__ cbh, const float* __restrict__ cbn,
    int* __restrict__ idxout, float* __restrict__ lout) {
    __shared__ float S[16][Kc];
    __shared__ float minv[16][8];
    __shared__ int   mini[16][8];

    int tid  = threadIdx.x;
    int wave = tid >> 5, lane = tid & 31;
    int half = lane >> 4, nlane = lane & 15;
    int b0   = blockIdx.x * 16;

    const _Float16* arow = zh + (size_t)(b0 + nlane) * Dd;
    v8f acc[8] = {};
#pragma unroll
    for (int k0 = 0; k0 < Dd; k0 += 32) {
        v16h a = load_a_frag_f16(arow, k0, half);
#pragma unroll
        for (int nt = 0; nt < 8; ++nt) {
            int n  = wave * 128 + nt * 16 + nlane;
            v16h b = load_b_frag_t(cbh, n, Dd, k0, half);
            acc[nt] = __builtin_amdgcn_wmma_f32_16x16x32_f16(
                false, a, false, b, (short)0, acc[nt], false, false);
        }
    }
#pragma unroll
    for (int nt = 0; nt < 8; ++nt) {
        int n    = wave * 128 + nt * 16 + nlane;
        float cn = cbn[n];
#pragma unroll
        for (int r = 0; r < 8; ++r) {
            int m = r + half * 8;
            S[m][n] = cn - 2.0f * acc[nt][r];
        }
    }
    __syncthreads();
    {   // 8 scanners per row, ascending order => first-min tiebreak like argmin
        int m = tid >> 3, seg = tid & 7;
        float bv = 3.4e38f; int bi = 0;
        for (int j = seg * 64; j < seg * 64 + 64; ++j) {
            float v = S[m][j];
            if (v < bv) { bv = v; bi = j; }
        }
        minv[m][seg] = bv; mini[m][seg] = bi;
    }
    __syncthreads();
    if (tid < 16) {
        int m = tid;
        float bv = minv[m][0]; int bi = mini[m][0];
        for (int s = 1; s < 8; ++s)
            if (minv[m][s] < bv) { bv = minv[m][s]; bi = mini[m][s]; }
        const float* zr = z + (size_t)(b0 + m) * Dd;
        float zn = 0.f;
        for (int j = 0; j < Dd; ++j) { float v = zr[j]; zn += v * v; }
        float d2 = zn + bv;                       // ||z - q||^2
        idxout[b0 + m] = bi;
        lout[b0 + m]   = (1.0f + BETA) * d2 / (float)Dd;
    }
}

// 5) decoder: out = codebook[idx] @ dec_W + dec_b ; fused img_out + MSE partials
__global__ __launch_bounds__(128) void k_decode(
    const _Float16* __restrict__ cbh, const int* __restrict__ idx,
    const _Float16* __restrict__ decWt, const float* __restrict__ decb,
    const float* __restrict__ img, float* __restrict__ out,
    float* __restrict__ partials) {
    __shared__ float red[128];

    int bid = blockIdx.x;
    int rt  = bid & 255;        // 256 row tiles
    int cbk = bid >> 8;         // 12 column blocks of 256
    int b0 = rt * 16, n_base = cbk * 256;

    int tid  = threadIdx.x;
    int wave = tid >> 5, lane = tid & 31;
    int half = lane >> 4, nlane = lane & 15;

    const _Float16* arow = cbh + (size_t)idx[b0 + nlane] * Dd;

    v8f acc[4] = {};
#pragma unroll
    for (int k0 = 0; k0 < Dd; k0 += 32) {
        v16h a = load_a_frag_f16(arow, k0, half);
#pragma unroll
        for (int nt = 0; nt < 4; ++nt) {
            int n  = n_base + wave * 64 + nt * 16 + nlane;
            v16h b = load_b_frag_t(decWt, n, Dd, k0, half);
            acc[nt] = __builtin_amdgcn_wmma_f32_16x16x32_f16(
                false, a, false, b, (short)0, acc[nt], false, false);
        }
    }
    float lsum = 0.f;
#pragma unroll
    for (int nt = 0; nt < 4; ++nt) {
        int col    = n_base + wave * 64 + nt * 16 + nlane;
        float bias = decb[col];
#pragma unroll
        for (int r = 0; r < 8; ++r) {
            int m    = r + half * 8;
            size_t o = (size_t)(b0 + m) * DinC + col;
            float v  = acc[nt][r] + bias;
            out[o]   = v;
            float d  = v - img[o];
            lsum += d * d;
        }
    }
    red[tid] = lsum;
    __syncthreads();
    if (tid == 0) {
        float s = 0.f;
        for (int i = 0; i < 128; ++i) s += red[i];   // fixed order
        partials[bid] = s;
    }
}

// 6) finalize: quant (fixed-order segment sums) + MSE -> loss
__global__ void k_finalize(const int* __restrict__ label,
                           const float* __restrict__ lvals,
                           const int* __restrict__ counts,
                           const float* __restrict__ partials, int nparts,
                           float* __restrict__ loss_out) {
    if (threadIdx.x != 0 || blockIdx.x != 0) return;
    float sums[Cc];
    for (int c = 0; c < Cc; ++c) sums[c] = 0.f;
    for (int b = 0; b < Bn; ++b) sums[label[b]] += lvals[b];
    float quant = 0.f, nact = 0.f;
    for (int c = 0; c < Cc; ++c)
        if (counts[c] > 0) { quant += sums[c] / (float)counts[c]; nact += 1.f; }
    if (nact < 1.f) nact = 1.f;
    quant /= nact;
    float mse = 0.f;
    for (int i = 0; i < nparts; ++i) mse += partials[i];
    mse /= (float)Bn * (float)DinC;
    loss_out[0] = mse + quant;
}

// ---------------------------------------------------------------------------
extern "C" void kernel_launch(void* const* d_in, const int* in_sizes, int n_in,
                              void* d_out, int out_size, void* d_ws, size_t ws_size,
                              hipStream_t stream) {
    const float* img   = (const float*)d_in[0];
    const int*   label = (const int*)d_in[1];
    const float* encW  = (const float*)d_in[2];
    const float* encb  = (const float*)d_in[3];
    const float* cb    = (const float*)d_in[4];
    const float* decW  = (const float*)d_in[5];
    const float* decb  = (const float*)d_in[6];
    float* out = (float*)d_out;          // [0]=loss, [1..]=img_out

    // workspace carve-up (256B aligned)
    char* w = (char*)d_ws;
    size_t off = 0;
    auto take = [&](size_t bytes) {
        char* p = w + off;
        off = (off + bytes + 255) & ~(size_t)255;
        return p;
    };
    int*      counts  = (int*)take(Cc * sizeof(int));
    int*      rowlist = (int*)take((size_t)Cc * Bn * sizeof(int));
    float*    z       = (float*)take((size_t)Bn * Dd * sizeof(float));
    _Float16* zh      = (_Float16*)take((size_t)Bn * Dd * sizeof(_Float16));
    float*    cbn     = (float*)take(Kc * sizeof(float));
    _Float16* cbh     = (_Float16*)take((size_t)Kc * Dd * sizeof(_Float16));
    int*      idxv    = (int*)take(Bn * sizeof(int));
    float*    lvals   = (float*)take(Bn * sizeof(float));
    float*    parts   = (float*)take(3072 * sizeof(float));
    _Float16* encWt   = (_Float16*)take((size_t)Cc * DinC * Dd * sizeof(_Float16));
    _Float16* decWt   = (_Float16*)take((size_t)DinC * Dd * sizeof(_Float16));
    (void)ws_size; (void)in_sizes; (void)n_in; (void)out_size;

    k_zero_counts<<<1, 32, 0, stream>>>(counts);
    k_scatter_rows<<<Bn / 256, 256, 0, stream>>>(label, counts, rowlist);
    k_prep_cb<<<Kc, 64, 0, stream>>>(cb, cbh, cbn);
    // enc_W[c]: [K=3072 x N=256] -> encWt[c]: [N=256][K=3072]
    k_transpose_f32_f16<<<dim3(Dd / 32, DinC / 32, Cc), 256, 0, stream>>>(
        encW, encWt, DinC, Dd, (size_t)DinC * Dd);
    // dec_W: [K=256 x N=3072] -> decWt: [N=3072][K=256]
    k_transpose_f32_f16<<<dim3(DinC / 32, Dd / 32, 1), 256, 0, stream>>>(
        decW, decWt, Dd, DinC, (size_t)Dd * DinC);

    k_encode<<<Cc * (Bn / 16), 128, 0, stream>>>(img, encWt, encb, counts,
                                                 rowlist, z, zh);
    k_vq<<<Bn / 16, 128, 0, stream>>>(z, zh, cbh, cbn, idxv, lvals);
    k_decode<<<(Bn / 16) * (DinC / 256), 128, 0, stream>>>(cbh, idxv, decWt,
                                                           decb, img, out + 1,
                                                           parts);
    k_finalize<<<1, 1, 0, stream>>>(label, lvals, counts, parts,
                                    (Bn / 16) * (DinC / 256), out);
}
